// ChebNetModel_29308856828499
// MI455X (gfx1250) — matrix operand order repaired and checked by reading, not compiled, not run on hardware
//
#include <hip/hip_runtime.h>
#include <hip/hip_bf16.h>

// ---------------------------------------------------------------------------
// ChebNet (3x ChebConv(K=4) + BN + ReLU, head Linear) for MI455X / gfx1250.
// Sparse propagations: fp32, hardware global_atomic_add_f32 scatter (L2-bound,
// whole working set < 192MB L2). Dense GEMMs: wave32 WMMA f16->f32 accumulate,
// fused bias + BatchNorm(eval) + ReLU epilogue. Weights are pre-packed once
// into f16 fragment-order so the WMMA loop loads B as contiguous b128s.
// ---------------------------------------------------------------------------

typedef __attribute__((ext_vector_type(16))) _Float16 v16h;
typedef __attribute__((ext_vector_type(8)))  float    v8f;

#define BN_EPS 1e-5f

// ------------------------- small utility kernels ---------------------------

__global__ void fill_zero_f(float* __restrict__ p, int n) {
    int t = blockIdx.x * blockDim.x + threadIdx.x;
    if (t < n) p[t] = 0.0f;
}

__global__ void fill_zero4(float4* __restrict__ p, int n4) {
    int t = blockIdx.x * blockDim.x + threadIdx.x;
    if (t < n4) p[t] = make_float4(0.f, 0.f, 0.f, 0.f);
}

__global__ void neg4(const float4* __restrict__ in, float4* __restrict__ out, int n4) {
    int t = blockIdx.x * blockDim.x + threadIdx.x;
    if (t < n4) {
        float4 v = in[t];
        out[t] = make_float4(-v.x, -v.y, -v.z, -v.w);
    }
}

// deg[src[e]] += 1
__global__ void deg_count(const int* __restrict__ src, float* __restrict__ deg, int E) {
    int t = blockIdx.x * blockDim.x + threadIdx.x;
    if (t < E) unsafeAtomicAdd(&deg[src[t]], 1.0f);
}

// deg -> dinv (in place):  dinv = deg>0 ? rsqrt(max(deg,1)) : 0
__global__ void make_dinv(float* __restrict__ deg, int N) {
    int t = blockIdx.x * blockDim.x + threadIdx.x;
    if (t < N) {
        float d = deg[t];
        deg[t] = (d > 0.0f) ? (1.0f / sqrtf(fmaxf(d, 1.0f))) : 0.0f;
    }
}

// w[e] = -dinv[src]*dinv[dst]
__global__ void edge_w(const int* __restrict__ src, const int* __restrict__ dst,
                       const float* __restrict__ dinv, float* __restrict__ w, int E) {
    int t = blockIdx.x * blockDim.x + threadIdx.x;
    if (t < E) w[t] = -dinv[src[t]] * dinv[dst[t]];
}

// out[dst[e]][:] += alpha * w[e] * x[src[e]][:]   (64 features, 16 thr/edge)
__global__ void prop_scatter(const float* __restrict__ x,
                             const int* __restrict__ src, const int* __restrict__ dst,
                             const float* __restrict__ w, float alpha,
                             float* __restrict__ out, int E) {
    unsigned t = blockIdx.x * blockDim.x + threadIdx.x;
    unsigned e = t >> 4;          // edge
    unsigned g = t & 15u;         // float4 group within 64 features
    if (e >= (unsigned)E) return;
    float we = alpha * w[e];
    int s = src[e];
    int d = dst[e];
    const float4 v = *(const float4*)(x + (size_t)s * 64 + g * 4);
    float* o = out + (size_t)d * 64 + g * 4;
    unsafeAtomicAdd(o + 0, we * v.x);
    unsafeAtomicAdd(o + 1, we * v.y);
    unsafeAtomicAdd(o + 2, we * v.z);
    unsafeAtomicAdd(o + 3, we * v.w);
}

// ------------------------- weight fragment packing -------------------------
// WMMA B fragment (32x16 f16): lane L -> (g = L/16, n-col = L%16),
//   element j (0..15): k = kk + j + 16*g (contiguous k order used by this code, same as A).
// Packed layout: half[frag][lane][j], frag = (kb*2 + kkIdx)*4 + nt.
// So in the GEMM loop each lane reads one contiguous 32-byte v16h.

// Cheb W: (4,64,64) f32 -> 32 frags * 32 lanes * 16 halves = 16384 halves
__global__ void pack_cheb_w(const float* __restrict__ W, _Float16* __restrict__ out) {
    int tid = blockIdx.x * blockDim.x + threadIdx.x;
    if (tid >= 16384) return;
    int j     = tid & 15;
    int lane  = (tid >> 4) & 31;
    int frag  = tid >> 9;          // 0..31
    int nt    = frag & 3;
    int kkIdx = (frag >> 2) & 1;
    int kb    = frag >> 3;
    int g  = lane >> 4;
    int mn = lane & 15;
    int k  = kkIdx * 32 + 16 * g + j;       // 0..63 within this cheb order
    int n  = nt * 16 + mn;                  // 0..63
    out[tid] = (_Float16)W[((size_t)kb * 64 + k) * 64 + n];
}

// head W: (64,16) f32 -> 2 frags * 32 lanes * 16 halves = 1024 halves
__global__ void pack_head_w(const float* __restrict__ W, _Float16* __restrict__ out) {
    int tid = blockIdx.x * blockDim.x + threadIdx.x;
    if (tid >= 1024) return;
    int j     = tid & 15;
    int lane  = (tid >> 4) & 31;
    int kkIdx = tid >> 9;          // 0..1
    int g  = lane >> 4;
    int mn = lane & 15;
    int k  = kkIdx * 32 + 16 * g + j;       // 0..63
    out[tid] = (_Float16)W[(size_t)k * 16 + mn];
}

// ------------------------------ WMMA GEMMs ---------------------------------
// A fragment (16x32 f16): lane -> (g = lane/16, m = lane%16)
//   element j: k = kk + j + 16*g (this code's contiguous k order, same as B)
// C/D (16x16 f32): element v -> row m = v + 8*g, col n = lane%16

__device__ __forceinline__ v16h load_a_frag(const float* __restrict__ arow,
                                            int kk, int g) {
    v16h a;
#pragma unroll
    for (int j = 0; j < 16; ++j) {
        a[j] = (_Float16)arow[kk + 16 * g + j];
    }
    return a;
}

// Fused: Y = relu( BN( T0@W0 + T1@W1 + T2@W2 + T3@W3 + bc ) )
__global__ __launch_bounds__(256)
void cheb_gemm_bn_relu(const float* __restrict__ T0, const float* __restrict__ T1,
                       const float* __restrict__ T2, const float* __restrict__ T3,
                       const _Float16* __restrict__ Wp, const float* __restrict__ bc,
                       const float* __restrict__ gam, const float* __restrict__ bet,
                       const float* __restrict__ rm,  const float* __restrict__ rv,
                       float* __restrict__ Y, int nTiles) {
    int wave = blockIdx.x * 8 + (threadIdx.x >> 5);
    if (wave >= nTiles) return;                 // wave-uniform: EXEC stays full
    int lane = threadIdx.x & 31;
    int g  = lane >> 4;
    int mn = lane & 15;
    int row0 = wave * 16;

    const float* Ts[4] = {T0, T1, T2, T3};
    v8f acc[4] = {};                            // 4 n-tiles of 16 cols each

#pragma unroll
    for (int kb = 0; kb < 4; ++kb) {
        const float* arow = Ts[kb] + (size_t)(row0 + mn) * 64;
#pragma unroll
        for (int kkIdx = 0; kkIdx < 2; ++kkIdx) {
            v16h a = load_a_frag(arow, kkIdx * 32, g);
#pragma unroll
            for (int nt = 0; nt < 4; ++nt) {
                int frag = (kb * 2 + kkIdx) * 4 + nt;
                v16h b = *(const v16h*)(Wp + ((size_t)frag * 32 + lane) * 16);
                acc[nt] = __builtin_amdgcn_wmma_f32_16x16x32_f16(
                    false, a, false, b, (short)0, acc[nt], false, false);
            }
        }
    }

    // epilogue: +bias, BatchNorm(eval), ReLU
#pragma unroll
    for (int nt = 0; nt < 4; ++nt) {
        int n = nt * 16 + mn;
        float scale = gam[n] / sqrtf(rv[n] + BN_EPS);
        float add   = bet[n] + (bc[n] - rm[n]) * scale;
#pragma unroll
        for (int v = 0; v < 8; ++v) {
            float val = acc[nt][v] * scale + add;
            val = fmaxf(val, 0.0f);
            Y[(size_t)(row0 + v + 8 * g) * 64 + n] = val;
        }
    }
}

// head: Y(N x 16) = X(N x 64) @ Wh(64 x 16) + Bh
__global__ __launch_bounds__(256)
void head_gemm(const float* __restrict__ X, const _Float16* __restrict__ Hp,
               const float* __restrict__ Bh, float* __restrict__ Y, int nTiles) {
    int wave = blockIdx.x * 8 + (threadIdx.x >> 5);
    if (wave >= nTiles) return;
    int lane = threadIdx.x & 31;
    int g  = lane >> 4;
    int mn = lane & 15;
    int row0 = wave * 16;

    const float* arow = X + (size_t)(row0 + mn) * 64;
    v8f acc = {};
#pragma unroll
    for (int kkIdx = 0; kkIdx < 2; ++kkIdx) {
        v16h a = load_a_frag(arow, kkIdx * 32, g);
        v16h b = *(const v16h*)(Hp + ((size_t)kkIdx * 32 + lane) * 16);
        acc = __builtin_amdgcn_wmma_f32_16x16x32_f16(
            false, a, false, b, (short)0, acc, false, false);
    }
    float bb = Bh[mn];
#pragma unroll
    for (int v = 0; v < 8; ++v)
        Y[(size_t)(row0 + v + 8 * g) * 16 + mn] = acc[v] + bb;
}

// ------------------------------- launcher ----------------------------------

static void run_layer(const float* X, const _Float16* Wp, const float* bc,
                      const float* gam, const float* bet,
                      const float* rm, const float* rv,
                      float* T1, float* T2, float* T3, float* Y,
                      const int* src, const int* dst, const float* w,
                      int N, int E, hipStream_t stream) {
    int n4 = N * 16;                               // N*64/4 float4s
    unsigned gb4 = (unsigned)((n4 + 255) / 256);
    unsigned gbE = (unsigned)(((size_t)E * 16 + 255) / 256);
    int nTiles = N / 16;
    unsigned gbG = (unsigned)((nTiles + 7) / 8);

    // T1 = L_hat @ X
    fill_zero4<<<gb4, 256, 0, stream>>>((float4*)T1, n4);
    prop_scatter<<<gbE, 256, 0, stream>>>(X, src, dst, w, 1.0f, T1, E);
    // T2 = 2*L_hat @ T1 - X
    neg4<<<gb4, 256, 0, stream>>>((const float4*)X, (float4*)T2, n4);
    prop_scatter<<<gbE, 256, 0, stream>>>(T1, src, dst, w, 2.0f, T2, E);
    // T3 = 2*L_hat @ T2 - T1
    neg4<<<gb4, 256, 0, stream>>>((const float4*)T1, (float4*)T3, n4);
    prop_scatter<<<gbE, 256, 0, stream>>>(T2, src, dst, w, 2.0f, T3, E);
    // Y = relu(BN(sum_k Tk @ Wk + bc))
    cheb_gemm_bn_relu<<<gbG, 256, 0, stream>>>(X, T1, T2, T3, Wp, bc,
                                               gam, bet, rm, rv, Y, nTiles);
}

extern "C" void kernel_launch(void* const* d_in, const int* in_sizes, int n_in,
                              void* d_out, int out_size, void* d_ws, size_t ws_size,
                              hipStream_t stream) {
    (void)n_in; (void)out_size; (void)ws_size;

    const float* x     = (const float*)d_in[0];
    const int*   ei    = (const int*)d_in[1];
    const float* W1    = (const float*)d_in[2];
    const float* bc1   = (const float*)d_in[3];
    const float* W2    = (const float*)d_in[4];
    const float* bc2   = (const float*)d_in[5];
    const float* W3    = (const float*)d_in[6];
    const float* bc3   = (const float*)d_in[7];
    const float* g1 = (const float*)d_in[8],  *be1 = (const float*)d_in[9];
    const float* rm1= (const float*)d_in[10], *rv1 = (const float*)d_in[11];
    const float* g2 = (const float*)d_in[12], *be2 = (const float*)d_in[13];
    const float* rm2= (const float*)d_in[14], *rv2 = (const float*)d_in[15];
    const float* g3 = (const float*)d_in[16], *be3 = (const float*)d_in[17];
    const float* rm3= (const float*)d_in[18], *rv3 = (const float*)d_in[19];
    const float* headW = (const float*)d_in[20];
    const float* headB = (const float*)d_in[21];

    const int N = in_sizes[0] / 64;   // 100000 (multiple of 16)
    const int E = in_sizes[1] / 2;    // 1600000
    const int* src = ei;
    const int* dst = ei + E;

    // workspace layout (floats, 256B-aligned chunks)
    float* ws   = (float*)d_ws;
    size_t Npad = ((size_t)N + 63) & ~(size_t)63;
    size_t Epad = ((size_t)E + 63) & ~(size_t)63;
    size_t NB   = (size_t)N * 64;
    float* deg  = ws;
    float* w    = ws + Npad;
    float* T1   = w + Epad;
    float* T2   = T1 + NB;
    float* T3   = T2 + NB;
    float* Y1   = T3 + NB;
    float* Y2   = Y1 + NB;
    _Float16* pW1 = (_Float16*)(Y2 + NB);   // 16384 halves each
    _Float16* pW2 = pW1 + 16384;
    _Float16* pW3 = pW2 + 16384;
    _Float16* pH  = pW3 + 16384;            // 1024 halves

    // pack weights to f16 fragment order (tiny, once per launch)
    pack_cheb_w<<<64, 256, 0, stream>>>(W1, pW1);
    pack_cheb_w<<<64, 256, 0, stream>>>(W2, pW2);
    pack_cheb_w<<<64, 256, 0, stream>>>(W3, pW3);
    pack_head_w<<<4, 256, 0, stream>>>(headW, pH);

    // graph normalization: deg -> dinv -> edge weights
    fill_zero_f<<<(N + 255) / 256, 256, 0, stream>>>(deg, N);
    deg_count  <<<(E + 255) / 256, 256, 0, stream>>>(src, deg, E);
    make_dinv  <<<(N + 255) / 256, 256, 0, stream>>>(deg, N);
    edge_w     <<<(E + 255) / 256, 256, 0, stream>>>(src, dst, deg, w, E);

    // three ChebConv + BN + ReLU layers
    run_layer(x,  pW1, bc1, g1, be1, rm1, rv1, T1, T2, T3, Y1, src, dst, w, N, E, stream);
    run_layer(Y1, pW2, bc2, g2, be2, rm2, rv2, T1, T2, T3, Y2, src, dst, w, N, E, stream);
    run_layer(Y2, pW3, bc3, g3, be3, rm3, rv3, T1, T2, T3, Y1, src, dst, w, N, E, stream);

    // head linear
    int nTiles = N / 16;
    head_gemm<<<(nTiles + 7) / 8, 256, 0, stream>>>(Y1, pH, headB,
                                                    (float*)d_out, nTiles);
}
